// HybridImageFittingModel_84834194030896
// MI455X (gfx1250) — compile-verified, matrix operand order checked by reading, not run on hardware
//
#include <hip/hip_runtime.h>

typedef __attribute__((ext_vector_type(16))) _Float16 v16h;
typedef __attribute__((ext_vector_type(2)))  __fp16   h2;     // matches __builtin_amdgcn_cvt_pkrtz return type
typedef __attribute__((ext_vector_type(8)))  float    v8f;
typedef __attribute__((ext_vector_type(2)))  float    f2;

#define NB_STATS 512
#define FRAG_ELEMS 36864                      // 3 layers * 3 types * 4 nt * 2 kt * 32 lanes * 16 elems
#define BN_OFF_BYTES (NB_STATS * 12 * 4)      // 24576
#define FRAG_OFF_BYTES (BN_OFF_BYTES + 64)    // 24640, 32B aligned for v16h loads

// ---------------- per-lane circuit helpers (fully unrolled -> registers, packed f32 math) ----

// state: st[k] = (re, im); rotation on pair -> v_pk_mul_f32 / v_pk_fma_f32
__device__ __forceinline__ void apply_ry(f2* st, int w, float c, float s) {
    const int bit = 1 << (5 - w);
#pragma unroll
    for (int base = 0; base < 64; ++base) {
        if (base & bit) continue;               // folds at compile time after unroll
        const int i0 = base, i1 = base | bit;
        const f2 a0 = st[i0], a1 = st[i1];
        st[i0] = a0 * c - a1 * s;
        st[i1] = a0 * s + a1 * c;
    }
}

__device__ __forceinline__ void apply_cnot(f2* st, int c, int t) {
    const int cb = 1 << (5 - c), tb = 1 << (5 - t);
#pragma unroll
    for (int k = 0; k < 64; ++k) {
        if ((k & cb) && !(k & tb)) {            // swap k <-> k|tb: pure register rename after unroll
            const int j = k | tb;
            const f2 tmp = st[k]; st[k] = st[j]; st[j] = tmp;
        }
    }
}

// ---------------- kernel 1: BatchNorm partial sums (h recomputed on the fly) ----------------

__global__ __launch_bounds__(256) void bn_stats_kernel(const float* __restrict__ x,
                                                       const float* __restrict__ w1,
                                                       const float* __restrict__ b1,
                                                       float* __restrict__ partials, int B) {
    __shared__ float sd[12][256];
    const int tid = threadIdx.x;
    float acc[12];
#pragma unroll
    for (int j = 0; j < 12; ++j) acc[j] = 0.f;
    for (int b = blockIdx.x * 256 + tid; b < B; b += NB_STATS * 256) {
        const float x0 = x[2 * b], x1 = x[2 * b + 1];
#pragma unroll
        for (int j = 0; j < 6; ++j) {
            const float h = fmaf(x0, w1[2 * j], fmaf(x1, w1[2 * j + 1], b1[j]));
            acc[j] += h;
            acc[6 + j] += h * h;
        }
    }
#pragma unroll
    for (int j = 0; j < 12; ++j) sd[j][tid] = acc[j];
    __syncthreads();
    for (int off = 128; off > 0; off >>= 1) {
        if (tid < off) {
#pragma unroll
            for (int j = 0; j < 12; ++j) sd[j][tid] += sd[j][tid + off];
        }
        __syncthreads();
    }
    if (tid < 12) partials[blockIdx.x * 12 + tid] = sd[tid][0];
}

// ---------------- kernel 2: finalize BN -> per-feature scale/shift ----------------

__global__ __launch_bounds__(256) void bn_finalize_kernel(const float* __restrict__ partials,
                                                          const float* __restrict__ gamma,
                                                          const float* __restrict__ beta,
                                                          float* __restrict__ bnp, int B) {
    __shared__ float sd[12][256];
    const int tid = threadIdx.x;
    float acc[12];
#pragma unroll
    for (int j = 0; j < 12; ++j) acc[j] = 0.f;
    for (int i = tid; i < NB_STATS; i += 256) {
#pragma unroll
        for (int j = 0; j < 12; ++j) acc[j] += partials[i * 12 + j];
    }
#pragma unroll
    for (int j = 0; j < 12; ++j) sd[j][tid] = acc[j];
    __syncthreads();
    for (int off = 128; off > 0; off >>= 1) {
        if (tid < off) {
#pragma unroll
            for (int j = 0; j < 12; ++j) sd[j][tid] += sd[j][tid + off];
        }
        __syncthreads();
    }
    if (tid < 6) {
        const float invB = 1.f / (float)B;
        const float mean = sd[tid][0] * invB;
        const float var  = sd[6 + tid][0] * invB - mean * mean;
        const float sc   = gamma[tid] * rsqrtf(var + 1e-5f);
        bnp[tid]     = sc;
        bnp[6 + tid] = beta[tid] - mean * sc;
    }
}

// ---------------- kernel 3: build layer unitaries (U^T) in WMMA B-fragment layout ----------------
// frag index t = ((layer*3+type)*4 + nt)*1024 + kt*512 + lane*16 + e
// B-frag element (lane, e) -> K = kt*32 + (e&7) + ((e>>3)<<4) + ((lane>>4)<<3), col j = nt*16 + (lane&15)
// value = U[j][K] (type 0: re, 1: im, 2: -im), U = Kron of 6 Rot matrices (wire 0 = MSB)

__global__ __launch_bounds__(256) void build_frag_kernel(const float* __restrict__ qw,
                                                         _Float16* __restrict__ frag) {
    const int t = blockIdx.x * 256 + threadIdx.x;
    if (t >= FRAG_ELEMS) return;
    const int e    = t & 15;
    const int lane = (t >> 4) & 31;
    const int kt   = (t >> 9) & 1;
    const int nt   = (t >> 10) & 3;
    const int lt   = t >> 12;            // 0..8
    const int type  = lt % 3;
    const int layer = lt / 3;

    const int j = nt * 16 + (lane & 15);
    const int K = kt * 32 + (e & 7) + ((e >> 3) << 4) + ((lane >> 4) << 3);

    float cr = 1.f, ci = 0.f;
#pragma unroll
    for (int w = 0; w < 6; ++w) {
        const float phi = qw[(layer * 6 + w) * 3 + 0];
        const float th  = qw[(layer * 6 + w) * 3 + 1];
        const float om  = qw[(layer * 6 + w) * 3 + 2];
        const float ct = cosf(0.5f * th), st = sinf(0.5f * th);
        const int bi = (j >> (5 - w)) & 1;   // row bit
        const int bk = (K >> (5 - w)) & 1;   // col bit
        float er, ei;
        if (bi == 0 && bk == 0)      { const float a = -0.5f * (phi + om); er = ct * cosf(a); ei = ct * sinf(a); }
        else if (bi == 0 && bk == 1) { const float a =  0.5f * (phi - om); er = -st * cosf(a); ei = -st * sinf(a); }
        else if (bi == 1 && bk == 0) { const float a = -0.5f * (phi - om); er = st * cosf(a); ei = st * sinf(a); }
        else                         { const float a =  0.5f * (phi + om); er = ct * cosf(a); ei = ct * sinf(a); }
        const float nr = cr * er - ci * ei;
        const float ni = cr * ei + ci * er;
        cr = nr; ci = ni;
    }
    const float v = (type == 0) ? cr : ((type == 1) ? ci : -ci);
    frag[t] = (_Float16)v;
}

// ---------------- kernel 4: fused main kernel ----------------
// 64 threads = 2 waves; each lane owns one sample's 64-amplitude complex state in registers.
// Rot layer = [32 samples, 64] x [64, 64] complex matmul via v_wmma_f32_16x16x32_f16.

__global__ __launch_bounds__(64) void hybrid_main_kernel(const float* __restrict__ x,
                                                         const float* __restrict__ w1,
                                                         const float* __restrict__ b1,
                                                         const float* __restrict__ w2,
                                                         const float* __restrict__ b2,
                                                         const float* __restrict__ bnp,
                                                         const _Float16* __restrict__ frag,
                                                         float* __restrict__ out, int B) {
    struct WaveLds {
        _Float16 inRe[32][64];   // f16 state, A-operand staging
        _Float16 inIm[32][64];
        f2       outSt[32][64];  // f32 (re,im) D-output staging
    };
    __shared__ WaveLds lds[2];

    const int tid  = threadIdx.x;
    const int lane = tid & 31;
    const int wv   = tid >> 5;
    const int hl   = lane >> 4;          // half-wave
    WaveLds& W = lds[wv];

    const int b  = blockIdx.x * 64 + tid;
    const int bc = (b < B) ? b : (B - 1);   // clamped load keeps EXEC uniform for WMMA

    const float x0 = x[2 * bc], x1 = x[2 * bc + 1];

    // normalized angles -> half-angle cos/sin; entangler angles via addition formulas
    float cI[6], sI[6];
#pragma unroll
    for (int j = 0; j < 6; ++j) {
        const float h  = fmaf(x0, w1[2 * j], fmaf(x1, w1[2 * j + 1], b1[j]));
        const float hn = fmaf(h, bnp[j], bnp[6 + j]);
        cI[j] = cosf(0.5f * hn);
        sI[j] = sinf(0.5f * hn);
    }
    float cS[5], sS[5];
#pragma unroll
    for (int i = 0; i < 5; ++i) {
        cS[i] = cI[i] * cI[i + 1] - sI[i] * sI[i + 1];
        sS[i] = sI[i] * cI[i + 1] + cI[i] * sI[i + 1];
    }

    f2 st[64];
#pragma unroll
    for (int k = 0; k < 64; ++k) { st[k].x = 0.f; st[k].y = 0.f; }
    st[0].x = 1.f;

    for (int layer = 0; layer < 3; ++layer) {
        // ---- per-sample VALU part: RY encoding + entangler (packed f32) ----
#pragma unroll
        for (int i = 0; i < 6; ++i) apply_ry(st, i, cI[i], sI[i]);
#pragma unroll
        for (int i = 0; i < 5; ++i) {
            apply_cnot(st, i, i + 1);
            apply_ry(st, i + 1, cS[i], sS[i]);
            apply_cnot(st, i + 1, i);
        }

        // ---- shared Rot layer: complex matmul with the layer's 64x64 unitary ----
#pragma unroll
        for (int k = 0; k < 64; k += 2) {    // packed f32->f16 conversion, 32-bit LDS stores
            const h2 pr = __builtin_amdgcn_cvt_pkrtz(st[k].x, st[k + 1].x);
            const h2 pi = __builtin_amdgcn_cvt_pkrtz(st[k].y, st[k + 1].y);
            *(h2*)&W.inRe[lane][k] = pr;
            *(h2*)&W.inIm[lane][k] = pi;
        }
        __syncthreads();

        // A fragments for both M tiles (state regs are dead here; ~64 VGPRs)
        v16h aRe[2][2], aIm[2][2];           // [mt][kt], 16x32 f16 A layout
#pragma unroll
        for (int mt = 0; mt < 2; ++mt) {
            const int m = mt * 16 + (lane & 15);
#pragma unroll
            for (int kt = 0; kt < 2; ++kt) {
#pragma unroll
                for (int e = 0; e < 16; ++e) {
                    const int K = kt * 32 + (e & 7) + ((e >> 3) << 4) + (hl << 3);
                    aRe[mt][kt][e] = W.inRe[m][K];
                    aIm[mt][kt][e] = W.inIm[m][K];
                }
            }
        }

        const _Float16* lf = frag + layer * 12288;   // [type][nt][kt][lane][16]
#pragma unroll
        for (int nt = 0; nt < 4; ++nt) {
            // B fragments loaded once, reused across both M tiles
            v16h bRe[2], bIm[2], bImN[2];
#pragma unroll
            for (int kt = 0; kt < 2; ++kt) {
                const _Float16* fb = lf + nt * 1024 + kt * 512 + lane * 16;
                bRe[kt]  = *(const v16h*)(fb);
                bIm[kt]  = *(const v16h*)(fb + 4096);
                bImN[kt] = *(const v16h*)(fb + 8192);   // pre-negated Uim
            }
            const int amp = nt * 16 + (lane & 15);
#pragma unroll
            for (int mt = 0; mt < 2; ++mt) {
                v8f cr = {}; v8f ci = {};
#pragma unroll
                for (int kt = 0; kt < 2; ++kt) {
                    cr = __builtin_amdgcn_wmma_f32_16x16x32_f16(false, aRe[mt][kt], false, bRe[kt],  (short)0, cr, false, false);
                    cr = __builtin_amdgcn_wmma_f32_16x16x32_f16(false, aIm[mt][kt], false, bImN[kt], (short)0, cr, false, false);
                    ci = __builtin_amdgcn_wmma_f32_16x16x32_f16(false, aRe[mt][kt], false, bIm[kt],  (short)0, ci, false, false);
                    ci = __builtin_amdgcn_wmma_f32_16x16x32_f16(false, aIm[mt][kt], false, bRe[kt],  (short)0, ci, false, false);
                }
                // D layout: VGPR v -> M = v + 8*hl, N = lane&15
                const int sb = mt * 16 + (hl << 3);
#pragma unroll
                for (int v = 0; v < 8; ++v) {
                    f2 t; t.x = cr[v]; t.y = ci[v];
                    W.outSt[sb + v][amp] = t;    // single b64 store
                }
            }
        }
        __syncthreads();
#pragma unroll
        for (int k = 0; k < 64; ++k) st[k] = W.outSt[lane][k];
        __syncthreads();
    }

    // ---- <Z_w> expectations fused with final linear layer ----
    float w2v[6];
#pragma unroll
    for (int w = 0; w < 6; ++w) w2v[w] = w2[w];
    float acc = b2[0];
#pragma unroll
    for (int k = 0; k < 64; ++k) {
        const f2 q = st[k] * st[k];
        const float p = q.x + q.y;
        float ck = 0.f;
#pragma unroll
        for (int w = 0; w < 6; ++w) ck += ((k >> (5 - w)) & 1) ? -w2v[w] : w2v[w];
        acc = fmaf(p, ck, acc);
    }
    if (b < B) out[b] = acc;
}

// ---------------- host launcher ----------------

extern "C" void kernel_launch(void* const* d_in, const int* in_sizes, int n_in,
                              void* d_out, int out_size, void* d_ws, size_t ws_size,
                              hipStream_t stream) {
    const float* x      = (const float*)d_in[0];
    const float* w1     = (const float*)d_in[1];
    const float* b1     = (const float*)d_in[2];
    const float* gamma  = (const float*)d_in[3];
    const float* beta   = (const float*)d_in[4];
    const float* qw     = (const float*)d_in[5];
    const float* w2     = (const float*)d_in[6];
    const float* b2     = (const float*)d_in[7];
    float* out = (float*)d_out;

    const int B = in_sizes[0] / 2;

    float*    partials = (float*)d_ws;
    float*    bnp      = (float*)((char*)d_ws + BN_OFF_BYTES);
    _Float16* frag     = (_Float16*)((char*)d_ws + FRAG_OFF_BYTES);

    bn_stats_kernel<<<NB_STATS, 256, 0, stream>>>(x, w1, b1, partials, B);
    bn_finalize_kernel<<<1, 256, 0, stream>>>(partials, gamma, beta, bnp, B);
    build_frag_kernel<<<(FRAG_ELEMS + 255) / 256, 256, 0, stream>>>(qw, frag);
    hybrid_main_kernel<<<(B + 63) / 64, 64, 0, stream>>>(x, w1, b1, w2, b2, bnp, frag, out, B);
}
